// Deformconv2dModify_87969520157239
// MI455X (gfx1250) — compile-verified, matrix operand order checked by reading, not run on hardware
//
#include <hip/hip_runtime.h>
#include <hip/hip_bf16.h>

typedef __attribute__((ext_vector_type(16))) _Float16 v16h;
typedef __attribute__((ext_vector_type(8)))  _Float16 v8h;
typedef __attribute__((ext_vector_type(8)))  float    v8f;

#define B_     8
#define CIN    128
#define COUT   256
#define HIMG   64
#define WIMG   64
#define HW     4096
#define TAPS   9
#define KDIM   (CIN * TAPS)      // 1152
#define KSTEPS (KDIM / 32)       // 36 (4 per tap j)
#define NTOT   (B_ * HW)         // 32768
#define BSTRIDE 40               // padded LDS row stride (halves) for B tile

// ---------------------------------------------------------------------------
// Kernel 0: convert + permute deformable-conv weights f32 -> f16.
// GEMM K is tap-major: k' = j*128 + c  (so a 32-wide K slab has constant j).
// wf16[o][k'] = dconv_w[o][c][j]
// ---------------------------------------------------------------------------
__global__ __launch_bounds__(256) void k_convert(const float* __restrict__ w,
                                                 _Float16* __restrict__ wf16) {
  int g = blockIdx.x * 256 + threadIdx.x;
  if (g >= COUT * KDIM) return;
  int o = g / KDIM;
  int r = g - o * KDIM;        // k' = j*128 + c
  int j = r >> 7;
  int c = r & 127;
  wf16[g] = (_Float16)w[(size_t)o * KDIM + c * TAPS + j];
}

// ---------------------------------------------------------------------------
// Kernel 1: offset conv.  One thread per (b, h, w), 18 accumulators.
// Weight index is wave-uniform -> scalar loads.   offbuf: [b][ch(18)][hw]
// ---------------------------------------------------------------------------
__global__ __launch_bounds__(256) void k_offset(const float* __restrict__ x,
                                                const float* __restrict__ ow,
                                                const float* __restrict__ ob,
                                                float* __restrict__ offbuf) {
  int g = blockIdx.x * 256 + threadIdx.x;   // 0 .. 32767
  int b = g >> 12;
  int n = g & (HW - 1);
  int h = n >> 6, w = n & 63;

  float acc[18];
#pragma unroll
  for (int ch = 0; ch < 18; ++ch) acc[ch] = ob[ch];

  for (int c = 0; c < CIN; ++c) {
    const float* xp = x + (size_t)(b * CIN + c) * HW;
#pragma unroll
    for (int j = 0; j < TAPS; ++j) {
      int yy = h - 1 + j / 3;
      int xx = w - 1 + j % 3;
      float xv = 0.f;
      if (yy >= 0 && yy < HIMG && xx >= 0 && xx < WIMG) xv = xp[yy * WIMG + xx];
      int base = c * TAPS + j;
#pragma unroll
      for (int ch = 0; ch < 18; ++ch) acc[ch] += xv * ow[ch * KDIM + base];
    }
  }
#pragma unroll
  for (int ch = 0; ch < 18; ++ch)
    offbuf[((size_t)(b * 18 + ch)) * HW + n] = acc[ch];
}

// ---------------------------------------------------------------------------
// Kernel 2: fused bilinear gather + WMMA GEMM.
// Block = (b, image row): computes y[b, 0..255, row, 0..63].
// K loop: 9 taps x 4 slabs of 32 channels. Per slab: 256 threads gather a
// 32(K) x 64(N) f16 tile (thread owns one n, 8 consecutive c; one
// ds_store_b128), double-buffered in LDS, one barrier per slab, then
// 8 waves x (2 M-tiles x 4 N-tiles) v_wmma_f32_16x16x32_f16.
// ---------------------------------------------------------------------------
__global__ __launch_bounds__(256) void k_dconv(const float* __restrict__ x,
                                               const float* __restrict__ offbuf,
                                               const _Float16* __restrict__ wf16,
                                               const float* __restrict__ bias,
                                               float* __restrict__ ybuf) {
  __shared__ float           s_w[576 * 4];                    // corner weights
  __shared__ unsigned short  s_idx[576 * 4];                  // corner indices
  __shared__ __align__(16) _Float16 s_b[2][64 * BSTRIDE];     // double-buffered B tile

  const int t   = threadIdx.x;
  const int row = blockIdx.x;           // h  (64 rows)
  const int b   = blockIdx.y;           // batch

  // ---- precompute sampling coefficients for this row's 64 positions x 9 taps
  for (int item = t; item < 576; item += 256) {
    int n = item / TAPS;                // w position (tile is one image row)
    int j = item % TAPS;
    int h = row, w = n;
    float dy = offbuf[((size_t)(b * 18 + j * 2    )) * HW + h * WIMG + w];
    float dx = offbuf[((size_t)(b * 18 + j * 2 + 1)) * HW + h * WIMG + w];
    float sy = (float)(h - 1 + j / 3) + dy;
    float sx = (float)(w - 1 + j % 3) + dx;
    float fy = floorf(sy), fx = floorf(sx);
    int y0 = (int)fy, x0 = (int)fx;
    float wy = sy - fy, wx = sx - fx;
    float cw[4] = {(1.f - wy) * (1.f - wx), (1.f - wy) * wx,
                   wy * (1.f - wx),         wy * wx};
    int cy[4] = {y0, y0, y0 + 1, y0 + 1};
    int cx[4] = {x0, x0 + 1, x0, x0 + 1};
#pragma unroll
    for (int ci = 0; ci < 4; ++ci) {
      bool v = (cy[ci] >= 0) && (cy[ci] < HIMG) && (cx[ci] >= 0) && (cx[ci] < WIMG);
      int yc = min(max(cy[ci], 0), HIMG - 1);
      int xc = min(max(cx[ci], 0), WIMG - 1);
      s_w[item * 4 + ci]   = v ? cw[ci] : 0.f;
      s_idx[item * 4 + ci] = (unsigned short)(yc * WIMG + xc);
    }
  }
  __syncthreads();

  const int lane = t & 31;
  const int wave = t >> 5;
  const int lrow = lane & 15;           // row-in-tile (M for A, N for B/C/D)
  const int lhi  = lane >> 4;           // K-chunk selector
  const int m0   = wave * 32;           // this wave's 2 M-tiles: m0, m0+16

  const int gn  = t & 63;               // gather: this thread's n position
  const int gk0 = (t >> 6) * 8;         // gather: k offset within 32-slab

  v8f acc[2][4] = {};

  for (int j = 0; j < TAPS; ++j) {
    // sampling params for (gn, j): constant across the 4 slabs of this tap
    int it4 = (gn * TAPS + j) * 4;
    float w0 = s_w[it4 + 0], w1 = s_w[it4 + 1];
    float w2 = s_w[it4 + 2], w3 = s_w[it4 + 3];
    int i0 = s_idx[it4 + 0], i1 = s_idx[it4 + 1];
    int i2 = s_idx[it4 + 2], i3 = s_idx[it4 + 3];

#pragma unroll
    for (int s4 = 0; s4 < 4; ++s4) {
      const int ks = j * 4 + s4;
      const int p  = ks & 1;

      // ---- A fragments (weights, K-contiguous; hot in L2) + prefetch next
      v16h af[2];
#pragma unroll
      for (int mi = 0; mi < 2; ++mi) {
        int r = m0 + mi * 16 + lrow;
        const _Float16* ap = wf16 + (size_t)r * KDIM + ks * 32 + lhi * 8;
        v8h lo = *(const v8h*)(ap);
        v8h hi = *(const v8h*)(ap + 16);
        af[mi] = __builtin_shufflevector(lo, hi, 0, 1, 2, 3, 4, 5, 6, 7,
                                                 8, 9, 10, 11, 12, 13, 14, 15);
        __builtin_prefetch(ap + 32, 0, 1);     // next slab (+64 B)
      }

      // ---- gather 8 consecutive channels for (gn, j) -> one ds_store_b128
      {
        const int c0 = s4 * 32 + gk0;
        const float* xp = x + (size_t)(b * CIN + c0) * HW;
        v8h gv;
#pragma unroll
        for (int i = 0; i < 8; ++i) {
          float v = w0 * xp[i0] + w1 * xp[i1] + w2 * xp[i2] + w3 * xp[i3];
          gv[i] = (_Float16)v;
          xp += HW;
        }
        *(v8h*)(&s_b[p][gn * BSTRIDE + gk0]) = gv;
      }
      __syncthreads();   // one barrier per slab (double-buffered tile)

      // ---- B fragments from LDS (n-major, per-lane chunk layout)
      v16h bf[4];
#pragma unroll
      for (int nt = 0; nt < 4; ++nt) {
        int n = nt * 16 + lrow;
        const _Float16* bp = &s_b[p][n * BSTRIDE + lhi * 8];
        v8h lo = *(const v8h*)(bp);
        v8h hi = *(const v8h*)(bp + 16);
        bf[nt] = __builtin_shufflevector(lo, hi, 0, 1, 2, 3, 4, 5, 6, 7,
                                                 8, 9, 10, 11, 12, 13, 14, 15);
      }

      // ---- 8 WMMAs: D = A x B + C
#pragma unroll
      for (int mi = 0; mi < 2; ++mi)
#pragma unroll
        for (int nt = 0; nt < 4; ++nt)
          acc[mi][nt] = __builtin_amdgcn_wmma_f32_16x16x32_f16(
              false, af[mi], false, bf[nt], (short)0, acc[mi][nt], false, false);
    }
  }

  // ---- epilogue: add bias, store pre-BN f32 tile
#pragma unroll
  for (int mi = 0; mi < 2; ++mi)
#pragma unroll
    for (int nt = 0; nt < 4; ++nt)
#pragma unroll
      for (int v = 0; v < 8; ++v) {
        int o = m0 + mi * 16 + v + lhi * 8;   // C/D layout: M = vgpr + 8*(lane/16)
        int n = nt * 16 + lrow;               // N = lane%16
        float val = acc[mi][nt][v] + bias[o];
        ybuf[((size_t)(b * COUT + o)) * HW + row * WIMG + n] = val;
      }
}

// ---------------------------------------------------------------------------
// Kernel 3: per-channel batch stats (deterministic tree reduction).
// stats[o] = mean, stats[256+o] = rsqrt(var + eps)   (biased var)
// ---------------------------------------------------------------------------
__global__ __launch_bounds__(256) void k_stats(const float* __restrict__ ybuf,
                                               float* __restrict__ stats) {
  __shared__ float rs[256], rq[256];
  int o = blockIdx.x, t = threadIdx.x;
  float s = 0.f, q = 0.f;
  for (int idx = t; idx < NTOT; idx += 256) {
    int b = idx >> 12;
    int n = idx & (HW - 1);
    float v = ybuf[((size_t)(b * COUT + o)) * HW + n];
    s += v;
    q += v * v;
  }
  rs[t] = s; rq[t] = q;
  __syncthreads();
  for (int off = 128; off > 0; off >>= 1) {
    if (t < off) { rs[t] += rs[t + off]; rq[t] += rq[t + off]; }
    __syncthreads();
  }
  if (t == 0) {
    float mean = rs[0] / (float)NTOT;
    float var  = rq[0] / (float)NTOT - mean * mean;
    stats[o]        = mean;
    stats[COUT + o] = rsqrtf(var + 1e-5f);
  }
}

// ---------------------------------------------------------------------------
// Kernel 4: BN affine + SiLU -> output [B][Cout][H][W] f32.
// ---------------------------------------------------------------------------
__global__ __launch_bounds__(256) void k_bnsilu(const float* __restrict__ ybuf,
                                                const float* __restrict__ stats,
                                                const float* __restrict__ gamma,
                                                const float* __restrict__ beta,
                                                float* __restrict__ out) {
  size_t e = (size_t)blockIdx.x * 256 + threadIdx.x;
  if (e >= (size_t)B_ * COUT * HW) return;
  int o = (int)((e >> 12) & (COUT - 1));   // (e / 4096) % 256
  float v = ybuf[e];
  float yh = (v - stats[o]) * stats[COUT + o] * gamma[o] + beta[o];
  out[e] = yh / (1.f + __expf(-yh));       // silu
}

// ---------------------------------------------------------------------------
extern "C" void kernel_launch(void* const* d_in, const int* in_sizes, int n_in,
                              void* d_out, int out_size, void* d_ws, size_t ws_size,
                              hipStream_t stream) {
  const float* x        = (const float*)d_in[0];
  const float* offset_w = (const float*)d_in[1];
  const float* offset_b = (const float*)d_in[2];
  const float* dconv_w  = (const float*)d_in[3];
  const float* dconv_b  = (const float*)d_in[4];
  const float* bn_gamma = (const float*)d_in[5];
  const float* bn_beta  = (const float*)d_in[6];
  float* out = (float*)d_out;

  char* ws = (char*)d_ws;
  float*    offbuf = (float*)ws;                       //  2,359,296 B
  _Float16* wf16   = (_Float16*)(ws + 2359296);        //    589,824 B
  float*    ybuf   = (float*)(ws + 2949120);           // 33,554,432 B
  float*    stats  = (float*)(ws + 36503552);          //      2,048 B

  k_convert<<<(COUT * KDIM + 255) / 256, 256, 0, stream>>>(dconv_w, wf16);
  k_offset<<<NTOT / 256, 256, 0, stream>>>(x, offset_w, offset_b, offbuf);
  k_dconv<<<dim3(HIMG, B_), 256, 0, stream>>>(x, offbuf, wf16, dconv_b, ybuf);
  k_stats<<<COUT, 256, 0, stream>>>(ybuf, stats);
  size_t tot = (size_t)B_ * COUT * HW;
  k_bnsilu<<<(unsigned)((tot + 255) / 256), 256, 0, stream>>>(ybuf, stats,
                                                              bn_gamma, bn_beta, out);
}
